// PairformerBlock_59176059405032
// MI455X (gfx1250) — compile-verified, matrix operand order checked by reading, not run on hardware
//
#include <hip/hip_runtime.h>

// ---------------------------------------------------------------------------
// CDNA5 (gfx1250) Pairformer block.  wave32, WMMA bf16 16x16x32, f32 accum.
// Double-buffered LDS staging; async global->LDS copies (GLOBAL_LOAD_ASYNC_TO_LDS).
// ---------------------------------------------------------------------------

typedef __bf16 bf16_t;
typedef __attribute__((ext_vector_type(16))) __bf16 v16bf;
typedef __attribute__((ext_vector_type(8)))  float  v8f;
typedef int v4i __attribute__((ext_vector_type(4)));

union Frag16 { v16bf v; uint4 q[2]; bf16_t e[16]; };
union AccF   { v8f   v; float f[8]; };
union Pack8  { uint4 u; bf16_t e[8]; };

__device__ __forceinline__ v8f wmma_bf16(v16bf a, v16bf b, v8f c) {
  return __builtin_amdgcn_wmma_f32_16x16x32_bf16(false, a, false, b, (short)0, c,
                                                 false, false);
}

#define LP 40  // LDS row pitch (bf16): 80B -> every 8-elem octet is 16B aligned

// ---- async global->LDS staging (gfx1250), guarded ----
#if defined(__has_builtin)
#if __has_builtin(__builtin_amdgcn_global_load_async_to_lds_b128)
#define HAVE_ASYNC_LDS 1
#endif
#endif
#ifndef HAVE_ASYNC_LDS
#define HAVE_ASYNC_LDS 0
#endif

__device__ __forceinline__ void stage16B(bf16_t* dst, const bf16_t* src) {
#if HAVE_ASYNC_LDS
  __builtin_amdgcn_global_load_async_to_lds_b128(
      (__attribute__((address_space(1))) v4i*)(uintptr_t)src,
      (__attribute__((address_space(3))) v4i*)(uint32_t)(uintptr_t)dst, 0, 0);
#else
  *(uint4*)dst = *(const uint4*)src;
#endif
}

__device__ __forceinline__ void stage_fence() {
#if HAVE_ASYNC_LDS
#if __has_builtin(__builtin_amdgcn_s_wait_asynccnt)
  __builtin_amdgcn_s_wait_asynccnt(0);
#else
  asm volatile("s_wait_asynccnt 0x0" ::: "memory");
#endif
#endif
}

__device__ __forceinline__ void load_frag16(const bf16_t* tile, Frag16& f) {
  int lane = threadIdx.x & 31;
  const bf16_t* p = tile + (lane & 15) * LP + ((lane >> 4) << 3);
  f.q[0] = *(const uint4*)(p);
  f.q[1] = *(const uint4*)(p + 16);
}

__device__ __forceinline__ float sigm(float x) { return 1.f / (1.f + __expf(-x)); }

__device__ __forceinline__ size_t mrow(int i, int j, int n, int trans) {
  return trans ? ((size_t)j * n + i) : ((size_t)i * n + j);
}

__device__ __forceinline__ float hmax16(float x) {
  for (int o = 8; o >= 1; o >>= 1) x = fmaxf(x, __shfl_xor(x, o, 32));
  return x;
}
__device__ __forceinline__ float hsum16(float x) {
  for (int o = 8; o >= 1; o >>= 1) x += __shfl_xor(x, o, 32);
  return x;
}

// ---------------------------------------------------------------------------
// Weight prep: W f32 [K][N] -> Wt bf16 [N][K]
// ---------------------------------------------------------------------------
__global__ __launch_bounds__(256) void k_wt(const float* __restrict__ W,
                                            bf16_t* __restrict__ Wt,
                                            int K, int Ncol) {
  int idx = blockIdx.x * 256 + threadIdx.x;
  if (idx < K * Ncol) {
    int k = idx / Ncol, nn = idx % Ncol;
    Wt[(size_t)nn * K + k] = (bf16_t)W[idx];
  }
}

// ---------------------------------------------------------------------------
// LayerNorm over last dim (f32 in -> bf16 out), one block per row.
// ---------------------------------------------------------------------------
__global__ __launch_bounds__(256) void k_ln_rows(const float* __restrict__ x,
                                                 const float* __restrict__ g,
                                                 const float* __restrict__ b,
                                                 bf16_t* __restrict__ out, int width) {
  __shared__ float red[256];
  size_t r = blockIdx.x;
  const float4* row4 = (const float4*)(x + r * (size_t)width);
  int w4 = width >> 2;
  float s = 0.f;
  for (int t = threadIdx.x; t < w4; t += 256) {
    float4 v = row4[t];
    s += v.x + v.y + v.z + v.w;
  }
  red[threadIdx.x] = s; __syncthreads();
  for (int o = 128; o >= 1; o >>= 1) { if (threadIdx.x < o) red[threadIdx.x] += red[threadIdx.x + o]; __syncthreads(); }
  float mean = red[0] / width; __syncthreads();
  float v = 0.f;
  for (int t = threadIdx.x; t < w4; t += 256) {
    float4 q = row4[t];
    float d0 = q.x - mean, d1 = q.y - mean, d2 = q.z - mean, d3 = q.w - mean;
    v += d0 * d0 + d1 * d1 + d2 * d2 + d3 * d3;
  }
  red[threadIdx.x] = v; __syncthreads();
  for (int o = 128; o >= 1; o >>= 1) { if (threadIdx.x < o) red[threadIdx.x] += red[threadIdx.x + o]; __syncthreads(); }
  float rstd = rsqrtf(red[0] / width + 1e-5f);
  for (int t = threadIdx.x; t < w4; t += 256) {
    float4 q = row4[t];
    int c = t * 4;
    Pack8 pk;
    pk.e[0] = (bf16_t)((q.x - mean) * rstd * g[c + 0] + b[c + 0]);
    pk.e[1] = (bf16_t)((q.y - mean) * rstd * g[c + 1] + b[c + 1]);
    pk.e[2] = (bf16_t)((q.z - mean) * rstd * g[c + 2] + b[c + 2]);
    pk.e[3] = (bf16_t)((q.w - mean) * rstd * g[c + 3] + b[c + 3]);
    *(uint2*)&out[r * (size_t)width + c] = *(uint2*)&pk.u;
  }
}

// LayerNorm across 128 channels of channel-major x (bf16 in -> bf16 row-major out)
__global__ __launch_bounds__(128) void k_ln_channels(const bf16_t* __restrict__ x_cm,
                                                     const float* __restrict__ g,
                                                     const float* __restrict__ b,
                                                     bf16_t* __restrict__ out, size_t R) {
  __shared__ float red[128];
  size_t r = blockIdx.x; int c = threadIdx.x;
  float v = (float)x_cm[(size_t)c * R + r];
  red[c] = v; __syncthreads();
  for (int o = 64; o >= 1; o >>= 1) { if (c < o) red[c] += red[c + o]; __syncthreads(); }
  float mean = red[0] / 128.f; __syncthreads();
  float d = v - mean; red[c] = d * d; __syncthreads();
  for (int o = 64; o >= 1; o >>= 1) { if (c < o) red[c] += red[c + o]; __syncthreads(); }
  float rstd = rsqrtf(red[0] / 128.f + 1e-5f);
  out[r * 128 + c] = (bf16_t)(d * rstd * g[c] + b[c]);
}

// Narrow per-row GEMV: out[r][h] = A[r,:] . W[:,h]  (H = 4 or 16)
__global__ __launch_bounds__(128) void k_rowvec(const bf16_t* __restrict__ A,
                                                const float* __restrict__ W,
                                                float* __restrict__ out, int K, int H) {
  __shared__ float red[128];
  size_t r = blockIdx.x;
  int t = threadIdx.x, h = t % H, seg = t / H;
  int nseg = 128 / H, L = K / nseg;
  const bf16_t* row = A + r * (size_t)K;
  float s = 0.f;
  for (int kk = 0; kk < L; kk++) {
    int k = seg * L + kk;
    s += (float)row[k] * W[(size_t)k * H + h];
  }
  red[t] = s; __syncthreads();
  if (t < H) {
    float tot = 0.f;
    for (int sg = 0; sg < nseg; sg++) tot += red[sg * H + t];
    out[r * (size_t)H + t] = tot;
  }
}

// ---------------------------------------------------------------------------
// GEMM: out_bf16 = act(A @ Wt^T (+bias)).  A bf16 [rows][K], Wt bf16 [Ncol][K].
// 64x64 macro-tile, 8 waves, double-buffered LDS, one barrier per K-step.
// ---------------------------------------------------------------------------
template <int ACT>
__global__ __launch_bounds__(256) void k_gemm_bf16(const bf16_t* __restrict__ A,
                                                   const bf16_t* __restrict__ Wt,
                                                   const float* __restrict__ bias,
                                                   bf16_t* __restrict__ out,
                                                   int K, int Ncol) {
  __shared__ __align__(16) bf16_t As[2][64 * LP];
  __shared__ __align__(16) bf16_t Bs[2][64 * LP];
  int m0 = blockIdx.x * 64, n0 = blockIdx.y * 64;
  int wid = threadIdx.x >> 5, wm = (wid & 3) * 16, wn = (wid >> 2) * 32;
  int tm = threadIdx.x >> 2, tg = (threadIdx.x & 3) * 8;
  int so = tm * LP + tg;
  const bf16_t* ga = A + (size_t)(m0 + tm) * K + tg;
  const bf16_t* gb = Wt + (size_t)(n0 + tm) * K + tg;
  AccF a0, a1;
  for (int i = 0; i < 8; i++) { a0.f[i] = 0.f; a1.f[i] = 0.f; }
  stage16B(&As[0][so], ga);
  stage16B(&Bs[0][so], gb);
  stage_fence();
  __syncthreads();
  int cur = 0;
  for (int k0 = 0; k0 < K; k0 += 32) {
    int nxt = cur ^ 1;
    if (k0 + 32 < K) {
      stage16B(&As[nxt][so], ga + k0 + 32);
      stage16B(&Bs[nxt][so], gb + k0 + 32);
      if (k0 + 64 < K) __builtin_prefetch(ga + k0 + 64, 0, 1);
    }
    Frag16 fa, fb0, fb1;
    load_frag16(&As[cur][wm * LP], fa);
    load_frag16(&Bs[cur][wn * LP], fb0);
    load_frag16(&Bs[cur][(wn + 16) * LP], fb1);
    a0.v = wmma_bf16(fa.v, fb0.v, a0.v);
    a1.v = wmma_bf16(fa.v, fb1.v, a1.v);
    stage_fence();
    __syncthreads();
    cur = nxt;
  }
  int lane = threadIdx.x & 31, nl = lane & 15, mh = (lane >> 4) * 8;
  for (int r = 0; r < 8; r++) {
    int m = m0 + wm + mh + r;
    int na = n0 + wn + nl, nb = na + 16;
    float v0 = a0.f[r], v1 = a1.f[r];
    if (bias) { v0 += bias[na]; v1 += bias[nb]; }
    if (ACT == 1) { v0 = sigm(v0); v1 = sigm(v1); }
    out[(size_t)m * Ncol + na] = (bf16_t)v0;
    out[(size_t)m * Ncol + nb] = (bf16_t)v1;
  }
}

// out_f32[m][n] += A@Wt^T (+bias)  (residual accumulation)
__global__ __launch_bounds__(256) void k_gemm_addf32(const bf16_t* __restrict__ A,
                                                     const bf16_t* __restrict__ Wt,
                                                     const float* __restrict__ bias,
                                                     float* __restrict__ out,
                                                     int K, int Ncol) {
  __shared__ __align__(16) bf16_t As[2][64 * LP];
  __shared__ __align__(16) bf16_t Bs[2][64 * LP];
  int m0 = blockIdx.x * 64, n0 = blockIdx.y * 64;
  int wid = threadIdx.x >> 5, wm = (wid & 3) * 16, wn = (wid >> 2) * 32;
  int tm = threadIdx.x >> 2, tg = (threadIdx.x & 3) * 8;
  int so = tm * LP + tg;
  const bf16_t* ga = A + (size_t)(m0 + tm) * K + tg;
  const bf16_t* gb = Wt + (size_t)(n0 + tm) * K + tg;
  AccF a0, a1;
  for (int i = 0; i < 8; i++) { a0.f[i] = 0.f; a1.f[i] = 0.f; }
  stage16B(&As[0][so], ga);
  stage16B(&Bs[0][so], gb);
  stage_fence();
  __syncthreads();
  int cur = 0;
  for (int k0 = 0; k0 < K; k0 += 32) {
    int nxt = cur ^ 1;
    if (k0 + 32 < K) {
      stage16B(&As[nxt][so], ga + k0 + 32);
      stage16B(&Bs[nxt][so], gb + k0 + 32);
      if (k0 + 64 < K) __builtin_prefetch(ga + k0 + 64, 0, 1);
    }
    Frag16 fa, fb0, fb1;
    load_frag16(&As[cur][wm * LP], fa);
    load_frag16(&Bs[cur][wn * LP], fb0);
    load_frag16(&Bs[cur][(wn + 16) * LP], fb1);
    a0.v = wmma_bf16(fa.v, fb0.v, a0.v);
    a1.v = wmma_bf16(fa.v, fb1.v, a1.v);
    stage_fence();
    __syncthreads();
    cur = nxt;
  }
  int lane = threadIdx.x & 31, nl = lane & 15, mh = (lane >> 4) * 8;
  for (int r = 0; r < 8; r++) {
    int m = m0 + wm + mh + r;
    int na = n0 + wn + nl, nb = na + 16;
    float v0 = a0.f[r], v1 = a1.f[r];
    if (bias) { v0 += bias[na]; v1 += bias[nb]; }
    out[(size_t)m * Ncol + na] += v0;
    out[(size_t)m * Ncol + nb] += v1;
  }
}

// Dual GEMM: o = act(A@W1t^T) * (A@W2t^T) [* mask[m]].  ACT: 1=sigmoid, 2=silu.
// CMAJOR=1 stores o[n*Rtot+m] bf16 (packed b128 along m).
template <int ACT, int CMAJOR>
__global__ __launch_bounds__(256) void k_dual_gate(const bf16_t* __restrict__ A,
                                                   const bf16_t* __restrict__ W1t,
                                                   const bf16_t* __restrict__ W2t,
                                                   const float* __restrict__ mask,
                                                   bf16_t* __restrict__ out,
                                                   int K, int Ncol, size_t Rtot) {
  __shared__ __align__(16) bf16_t As[2][64 * LP];
  __shared__ __align__(16) bf16_t B1[2][64 * LP];
  __shared__ __align__(16) bf16_t B2[2][64 * LP];
  int m0 = blockIdx.x * 64, n0 = blockIdx.y * 64;
  int wid = threadIdx.x >> 5, wm = (wid & 3) * 16, wn = (wid >> 2) * 32;
  int tm = threadIdx.x >> 2, tg = (threadIdx.x & 3) * 8;
  int so = tm * LP + tg;
  const bf16_t* ga = A + (size_t)(m0 + tm) * K + tg;
  const bf16_t* g1 = W1t + (size_t)(n0 + tm) * K + tg;
  const bf16_t* g2 = W2t + (size_t)(n0 + tm) * K + tg;
  AccF p0, p1, q0, q1;
  for (int i = 0; i < 8; i++) { p0.f[i] = 0.f; p1.f[i] = 0.f; q0.f[i] = 0.f; q1.f[i] = 0.f; }
  stage16B(&As[0][so], ga);
  stage16B(&B1[0][so], g1);
  stage16B(&B2[0][so], g2);
  stage_fence();
  __syncthreads();
  int cur = 0;
  for (int k0 = 0; k0 < K; k0 += 32) {
    int nxt = cur ^ 1;
    if (k0 + 32 < K) {
      stage16B(&As[nxt][so], ga + k0 + 32);
      stage16B(&B1[nxt][so], g1 + k0 + 32);
      stage16B(&B2[nxt][so], g2 + k0 + 32);
      if (k0 + 64 < K) __builtin_prefetch(ga + k0 + 64, 0, 1);
    }
    Frag16 fa, f1a, f1b, f2a, f2b;
    load_frag16(&As[cur][wm * LP], fa);
    load_frag16(&B1[cur][wn * LP], f1a);
    load_frag16(&B1[cur][(wn + 16) * LP], f1b);
    load_frag16(&B2[cur][wn * LP], f2a);
    load_frag16(&B2[cur][(wn + 16) * LP], f2b);
    p0.v = wmma_bf16(fa.v, f1a.v, p0.v);
    p1.v = wmma_bf16(fa.v, f1b.v, p1.v);
    q0.v = wmma_bf16(fa.v, f2a.v, q0.v);
    q1.v = wmma_bf16(fa.v, f2b.v, q1.v);
    stage_fence();
    __syncthreads();
    cur = nxt;
  }
  int lane = threadIdx.x & 31, nl = lane & 15, mh = (lane >> 4) * 8;
  int mb = m0 + wm + mh;
  int na = n0 + wn + nl, nb = na + 16;
  Pack8 o0, o1;
  for (int r = 0; r < 8; r++) {
    float g0 = (ACT == 2) ? (p0.f[r] * sigm(p0.f[r])) : sigm(p0.f[r]);
    float g1v = (ACT == 2) ? (p1.f[r] * sigm(p1.f[r])) : sigm(p1.f[r]);
    float v0 = g0 * q0.f[r], v1 = g1v * q1.f[r];
    if (mask) { float mk = mask[mb + r]; v0 *= mk; v1 *= mk; }
    if (CMAJOR) {
      o0.e[r] = (bf16_t)v0; o1.e[r] = (bf16_t)v1;
    } else {
      out[(size_t)(mb + r) * Ncol + na] = (bf16_t)v0;
      out[(size_t)(mb + r) * Ncol + nb] = (bf16_t)v1;
    }
  }
  if (CMAJOR) {
    *(uint4*)&out[(size_t)na * Rtot + mb] = o0.u;
    *(uint4*)&out[(size_t)nb * Rtot + mb] = o1.u;
  }
}

// Final triangle-mul GEMM: Z[m][n] += (A1@W1t^T) * sigmoid(A2@W2t^T)
__global__ __launch_bounds__(256) void k_dual_final(const bf16_t* __restrict__ A1,
                                                    const bf16_t* __restrict__ A2,
                                                    const bf16_t* __restrict__ W1t,
                                                    const bf16_t* __restrict__ W2t,
                                                    float* __restrict__ Z,
                                                    int K, int Ncol) {
  __shared__ __align__(16) bf16_t As1[2][64 * LP];
  __shared__ __align__(16) bf16_t As2[2][64 * LP];
  __shared__ __align__(16) bf16_t B1[2][64 * LP];
  __shared__ __align__(16) bf16_t B2[2][64 * LP];
  int m0 = blockIdx.x * 64, n0 = blockIdx.y * 64;
  int wid = threadIdx.x >> 5, wm = (wid & 3) * 16, wn = (wid >> 2) * 32;
  int tm = threadIdx.x >> 2, tg = (threadIdx.x & 3) * 8;
  int so = tm * LP + tg;
  const bf16_t* ga1 = A1 + (size_t)(m0 + tm) * K + tg;
  const bf16_t* ga2 = A2 + (size_t)(m0 + tm) * K + tg;
  const bf16_t* g1 = W1t + (size_t)(n0 + tm) * K + tg;
  const bf16_t* g2 = W2t + (size_t)(n0 + tm) * K + tg;
  AccF p0, p1, q0, q1;
  for (int i = 0; i < 8; i++) { p0.f[i] = 0.f; p1.f[i] = 0.f; q0.f[i] = 0.f; q1.f[i] = 0.f; }
  stage16B(&As1[0][so], ga1);
  stage16B(&As2[0][so], ga2);
  stage16B(&B1[0][so], g1);
  stage16B(&B2[0][so], g2);
  stage_fence();
  __syncthreads();
  int cur = 0;
  for (int k0 = 0; k0 < K; k0 += 32) {
    int nxt = cur ^ 1;
    if (k0 + 32 < K) {
      stage16B(&As1[nxt][so], ga1 + k0 + 32);
      stage16B(&As2[nxt][so], ga2 + k0 + 32);
      stage16B(&B1[nxt][so], g1 + k0 + 32);
      stage16B(&B2[nxt][so], g2 + k0 + 32);
    }
    Frag16 fa1, fa2, f1a, f1b, f2a, f2b;
    load_frag16(&As1[cur][wm * LP], fa1);
    load_frag16(&As2[cur][wm * LP], fa2);
    load_frag16(&B1[cur][wn * LP], f1a);
    load_frag16(&B1[cur][(wn + 16) * LP], f1b);
    load_frag16(&B2[cur][wn * LP], f2a);
    load_frag16(&B2[cur][(wn + 16) * LP], f2b);
    p0.v = wmma_bf16(fa1.v, f1a.v, p0.v);
    p1.v = wmma_bf16(fa1.v, f1b.v, p1.v);
    q0.v = wmma_bf16(fa2.v, f2a.v, q0.v);
    q1.v = wmma_bf16(fa2.v, f2b.v, q1.v);
    stage_fence();
    __syncthreads();
    cur = nxt;
  }
  int lane = threadIdx.x & 31, nl = lane & 15, mh = (lane >> 4) * 8;
  for (int r = 0; r < 8; r++) {
    int m = m0 + wm + mh + r;
    int na = n0 + wn + nl, nb = na + 16;
    Z[(size_t)m * Ncol + na] += p0.f[r] * sigm(q0.f[r]);
    Z[(size_t)m * Ncol + nb] += p1.f[r] * sigm(q1.f[r]);
  }
}

// ---------------------------------------------------------------------------
// Triangle einsum, 128 channel-major 384^3 GEMMs.
// OUTG=1: X[i,j] = sum_k A[i,k]B[j,k]   OUTG=0: X[i,j] = sum_k A[k,i]B[k,j]
// ---------------------------------------------------------------------------
template <int OUTG>
__global__ __launch_bounds__(256) void k_tri_einsum(const bf16_t* __restrict__ a_cm,
                                                    const bf16_t* __restrict__ b_cm,
                                                    bf16_t* __restrict__ x_cm,
                                                    int n) {
  __shared__ __align__(16) bf16_t As[2][64 * LP];
  __shared__ __align__(16) bf16_t Bs[2][64 * LP];
  int c = blockIdx.z;
  const bf16_t* A = a_cm + (size_t)c * n * n;
  const bf16_t* Bm = b_cm + (size_t)c * n * n;
  bf16_t* X = x_cm + (size_t)c * n * n;
  int m0 = blockIdx.x * 64, n0 = blockIdx.y * 64;
  int wid = threadIdx.x >> 5, wm = (wid & 3) * 16, wn = (wid >> 2) * 32;

  auto stage = [&](int k0, int buf) {
    if (OUTG) {
      int tm = threadIdx.x >> 2, tg = (threadIdx.x & 3) * 8;
      stage16B(&As[buf][tm * LP + tg], &A[(size_t)(m0 + tm) * n + k0 + tg]);
      stage16B(&Bs[buf][tm * LP + tg], &Bm[(size_t)(n0 + tm) * n + k0 + tg]);
    } else {
      // transposed: vector load 8 consecutive m/n at fixed k, scatter into LDS
      int kk = threadIdx.x >> 3, mg = (threadIdx.x & 7) * 8;
      Pack8 va, vb;
      va.u = *(const uint4*)&A[(size_t)(k0 + kk) * n + m0 + mg];
      vb.u = *(const uint4*)&Bm[(size_t)(k0 + kk) * n + n0 + mg];
      for (int j = 0; j < 8; j++) {
        As[buf][(mg + j) * LP + kk] = va.e[j];
        Bs[buf][(mg + j) * LP + kk] = vb.e[j];
      }
    }
  };

  AccF a0, a1;
  for (int i = 0; i < 8; i++) { a0.f[i] = 0.f; a1.f[i] = 0.f; }
  stage(0, 0);
  stage_fence();
  __syncthreads();
  int cur = 0;
  for (int k0 = 0; k0 < n; k0 += 32) {
    int nxt = cur ^ 1;
    if (k0 + 32 < n) stage(k0 + 32, nxt);
    Frag16 fa, fb0, fb1;
    load_frag16(&As[cur][wm * LP], fa);
    load_frag16(&Bs[cur][wn * LP], fb0);
    load_frag16(&Bs[cur][(wn + 16) * LP], fb1);
    a0.v = wmma_bf16(fa.v, fb0.v, a0.v);
    a1.v = wmma_bf16(fa.v, fb1.v, a1.v);
    stage_fence();
    __syncthreads();
    cur = nxt;
  }
  int lane = threadIdx.x & 31, nl = lane & 15, mh = (lane >> 4) * 8;
  for (int r = 0; r < 8; r++) {
    int m = m0 + wm + mh + r;
    X[(size_t)m * n + n0 + wn + nl] = (bf16_t)a0.f[r];
    X[(size_t)m * n + n0 + wn + 16 + nl] = (bf16_t)a1.f[r];
  }
}

// ---------------------------------------------------------------------------
// Triangle attention: flash attention per (i, head), one wave per 16-row tile.
// ---------------------------------------------------------------------------
__global__ __launch_bounds__(32) void k_tri_flash(const bf16_t* __restrict__ q,
                                                  const bf16_t* __restrict__ kbuf,
                                                  const bf16_t* __restrict__ vbuf,
                                                  const bf16_t* __restrict__ gate,
                                                  const float* __restrict__ bias4,
                                                  const float* __restrict__ mask,
                                                  bf16_t* __restrict__ out,
                                                  int n, int trans) {
  __shared__ __align__(16) bf16_t Qs[16 * LP];
  __shared__ __align__(16) bf16_t Ks[32 * LP];
  __shared__ __align__(16) bf16_t Vt[32 * LP];
  __shared__ __align__(16) bf16_t Ps[16 * LP];
  int j0 = blockIdx.x * 16, h = blockIdx.y, i = blockIdx.z;
  int lane = threadIdx.x, nl = lane & 15, mh = (lane >> 4) * 8;
  const float scale = 0.17677669529663687f;  // 1/sqrt(32)

  for (int s = lane; s < 64; s += 32) {
    int jj = s >> 2, dg = (s & 3) * 8;
    *(uint4*)&Qs[jj * LP + dg] =
        *(const uint4*)&q[mrow(i, j0 + jj, n, trans) * 128 + h * 32 + dg];
  }
  __syncthreads();
  Frag16 qf; load_frag16(Qs, qf);

  AccF Oa, Ob; float mst[8], lst[8];
  for (int r = 0; r < 8; r++) { Oa.f[r] = 0.f; Ob.f[r] = 0.f; mst[r] = -3e38f; lst[r] = 0.f; }

  for (int k0 = 0; k0 < n; k0 += 32) {
    for (int s = lane; s < 128; s += 32) {
      int kk = s >> 2, dg = (s & 3) * 8;
      size_t base = mrow(i, k0 + kk, n, trans) * 128 + h * 32 + dg;
      *(uint4*)&Ks[kk * LP + dg] = *(const uint4*)&kbuf[base];
      Pack8 vv; vv.u = *(const uint4*)&vbuf[base];
      for (int j = 0; j < 8; j++) Vt[(dg + j) * LP + kk] = vv.e[j];
    }
    __syncthreads();
    Frag16 fb0, fb1; load_frag16(Ks, fb0); load_frag16(Ks + 16 * LP, fb1);
    AccF S1, S2;
    for (int r = 0; r < 8; r++) { S1.f[r] = 0.f; S2.f[r] = 0.f; }
    S1.v = wmma_bf16(qf.v, fb0.v, S1.v);
    S2.v = wmma_bf16(qf.v, fb1.v, S2.v);

    for (int r = 0; r < 8; r++) {
      int jj = mh + r, kk1 = k0 + nl, kk2 = k0 + 16 + nl;
      float s1 = S1.f[r] * scale + bias4[mrow(j0 + jj, kk1, n, trans) * 4 + h];
      float s2 = S2.f[r] * scale + bias4[mrow(j0 + jj, kk2, n, trans) * 4 + h];
      if (mask[mrow(i, kk1, n, trans)] <= 0.f) s1 = -1e9f;
      if (mask[mrow(i, kk2, n, trans)] <= 0.f) s2 = -1e9f;
      float mnew = fmaxf(mst[r], hmax16(fmaxf(s1, s2)));
      float p1 = __expf(s1 - mnew), p2 = __expf(s2 - mnew);
      float rs = hsum16(p1 + p2);
      float corr = __expf(mst[r] - mnew);
      lst[r] = lst[r] * corr + rs;
      mst[r] = mnew;
      Oa.f[r] *= corr; Ob.f[r] *= corr;
      Ps[jj * LP + nl] = (bf16_t)p1;
      Ps[jj * LP + 16 + nl] = (bf16_t)p2;
    }
    __syncthreads();
    Frag16 pf, vf0, vf1;
    load_frag16(Ps, pf);
    load_frag16(Vt, vf0);
    load_frag16(Vt + 16 * LP, vf1);
    Oa.v = wmma_bf16(pf.v, vf0.v, Oa.v);
    Ob.v = wmma_bf16(pf.v, vf1.v, Ob.v);
    __syncthreads();
  }
  for (int r = 0; r < 8; r++) {
    float inv = lst[r] > 0.f ? 1.f / lst[r] : 0.f;
    size_t ro = mrow(i, j0 + mh + r, n, trans) * 128 + h * 32;
    out[ro + nl]      = (bf16_t)(Oa.f[r] * inv * (float)gate[ro + nl]);
    out[ro + 16 + nl] = (bf16_t)(Ob.f[r] * inv * (float)gate[ro + 16 + nl]);
  }
}

// ---------------------------------------------------------------------------
// Pair-bias single attention over s: 16 heads, D=24 (zero-padded to 32).
// ---------------------------------------------------------------------------
__global__ __launch_bounds__(32) void k_s_flash(const bf16_t* __restrict__ qkv,
                                                const float* __restrict__ bias16,
                                                const float* __restrict__ mask,
                                                const bf16_t* __restrict__ gate,
                                                bf16_t* __restrict__ out, int n) {
  __shared__ __align__(16) bf16_t Qs[16 * LP];
  __shared__ __align__(16) bf16_t Ks[32 * LP];
  __shared__ __align__(16) bf16_t Vt[32 * LP];
  __shared__ __align__(16) bf16_t Ps[16 * LP];
  const int D = 24, QW = 1152;
  int i0 = blockIdx.x * 16, h = blockIdx.y;
  int lane = threadIdx.x, nl = lane & 15, mh = (lane >> 4) * 8;
  const float scale = 0.20412414523193154f;  // 1/sqrt(24)
  const uint4 zero4 = {0, 0, 0, 0};

  for (int s = lane; s < 64; s += 32) {
    int ii = s >> 2, dg = (s & 3) * 8;
    *(uint4*)&Qs[ii * LP + dg] =
        (dg < D) ? *(const uint4*)&qkv[(size_t)(i0 + ii) * QW + h * D + dg] : zero4;
  }
  __syncthreads();
  Frag16 qf; load_frag16(Qs, qf);

  AccF Oa, Ob; float mst[8], lst[8];
  for (int r = 0; r < 8; r++) { Oa.f[r] = 0.f; Ob.f[r] = 0.f; mst[r] = -3e38f; lst[r] = 0.f; }

  for (int j0 = 0; j0 < n; j0 += 32) {
    for (int s = lane; s < 128; s += 32) {
      int kk = s >> 2, dg = (s & 3) * 8;
      size_t base = (size_t)(j0 + kk) * QW + h * D + dg;
      Pack8 vv;
      if (dg < D) {
        *(uint4*)&Ks[kk * LP + dg] = *(const uint4*)&qkv[base + 384];
        vv.u = *(const uint4*)&qkv[base + 768];
      } else {
        *(uint4*)&Ks[kk * LP + dg] = zero4;
        vv.u = zero4;
      }
      for (int j = 0; j < 8; j++) Vt[(dg + j) * LP + kk] = vv.e[j];
    }
    __syncthreads();
    Frag16 fb0, fb1; load_frag16(Ks, fb0); load_frag16(Ks + 16 * LP, fb1);
    AccF S1, S2;
    for (int r = 0; r < 8; r++) { S1.f[r] = 0.f; S2.f[r] = 0.f; }
    S1.v = wmma_bf16(qf.v, fb0.v, S1.v);
    S2.v = wmma_bf16(qf.v, fb1.v, S2.v);

    for (int r = 0; r < 8; r++) {
      int ii = i0 + mh + r, jj1 = j0 + nl, jj2 = j0 + 16 + nl;
      size_t r1 = (size_t)ii * n + jj1, r2 = (size_t)ii * n + jj2;
      float s1 = S1.f[r] * scale + (mask[r1] > 0.f ? bias16[r1 * 16 + h] : -1e9f);
      float s2 = S2.f[r] * scale + (mask[r2] > 0.f ? bias16[r2 * 16 + h] : -1e9f);
      float mnew = fmaxf(mst[r], hmax16(fmaxf(s1, s2)));
      float p1 = __expf(s1 - mnew), p2 = __expf(s2 - mnew);
      float rs = hsum16(p1 + p2);
      float corr = __expf(mst[r] - mnew);
      lst[r] = lst[r] * corr + rs;
      mst[r] = mnew;
      Oa.f[r] *= corr; Ob.f[r] *= corr;
      Ps[(mh + r) * LP + nl] = (bf16_t)p1;
      Ps[(mh + r) * LP + 16 + nl] = (bf16_t)p2;
    }
    __syncthreads();
    Frag16 pf, vf0, vf1;
    load_frag16(Ps, pf);
    load_frag16(Vt, vf0);
    load_frag16(Vt + 16 * LP, vf1);
    Oa.v = wmma_bf16(pf.v, vf0.v, Oa.v);
    Ob.v = wmma_bf16(pf.v, vf1.v, Ob.v);
    __syncthreads();
  }
  for (int r = 0; r < 8; r++) {
    float inv = lst[r] > 0.f ? 1.f / lst[r] : 0.f;
    int ii = i0 + mh + r;
    size_t o0 = (size_t)ii * 384 + h * D + nl;
    out[o0] = (bf16_t)(Oa.f[r] * inv * (float)gate[o0]);
    if (16 + nl < D) {
      size_t o1 = (size_t)ii * 384 + h * D + 16 + nl;
      out[o1] = (bf16_t)(Ob.f[r] * inv * (float)gate[o1]);
    }
  }
}

// ---------------------------------------------------------------------------
// Host-side orchestration
// ---------------------------------------------------------------------------
static void tri_mul_pass(float* Z, const float* mask,
                         const float* ln_in_g, const float* ln_in_b,
                         const bf16_t* wagt, const bf16_t* wapt,
                         const bf16_t* wbgt, const bf16_t* wbpt,
                         const float* ln_out_g, const float* ln_out_b,
                         const bf16_t* wzt, const bf16_t* wgt,
                         bf16_t* zn, bf16_t* a_cm, bf16_t* b_cm, bf16_t* x_cm,
                         bf16_t* xn, int outgoing, hipStream_t st) {
  const int Nn = 384; const size_t R = (size_t)Nn * Nn;
  k_ln_rows<<<dim3((unsigned)R), 256, 0, st>>>(Z, ln_in_g, ln_in_b, zn, 128);
  k_dual_gate<1, 1><<<dim3((unsigned)(R / 64), 2), 256, 0, st>>>(zn, wagt, wapt, mask, a_cm, 128, 128, R);
  k_dual_gate<1, 1><<<dim3((unsigned)(R / 64), 2), 256, 0, st>>>(zn, wbgt, wbpt, mask, b_cm, 128, 128, R);
  if (outgoing)
    k_tri_einsum<1><<<dim3(6, 6, 128), 256, 0, st>>>(a_cm, b_cm, x_cm, Nn);
  else
    k_tri_einsum<0><<<dim3(6, 6, 128), 256, 0, st>>>(a_cm, b_cm, x_cm, Nn);
  k_ln_channels<<<dim3((unsigned)R), 128, 0, st>>>(x_cm, ln_out_g, ln_out_b, xn, R);
  k_dual_final<<<dim3((unsigned)(R / 64), 2), 256, 0, st>>>(xn, zn, wzt, wgt, Z, 128, 128);
}

static void tri_att_pass(float* Z, const float* mask,
                         const float* ln_g, const float* ln_b,
                         const bf16_t* wqt, const bf16_t* wkt, const bf16_t* wvt,
                         const float* w_b, const bf16_t* wgatet,
                         const bf16_t* wot, const float* b_o,
                         bf16_t* zn, bf16_t* qb, bf16_t* kb, bf16_t* vb, bf16_t* gb,
                         float* bias4, int trans, hipStream_t st) {
  const int Nn = 384; const size_t R = (size_t)Nn * Nn;
  k_ln_rows<<<dim3((unsigned)R), 256, 0, st>>>(Z, ln_g, ln_b, zn, 128);
  k_gemm_bf16<0><<<dim3((unsigned)(R / 64), 2), 256, 0, st>>>(zn, wqt, nullptr, qb, 128, 128);
  k_gemm_bf16<0><<<dim3((unsigned)(R / 64), 2), 256, 0, st>>>(zn, wkt, nullptr, kb, 128, 128);
  k_gemm_bf16<0><<<dim3((unsigned)(R / 64), 2), 256, 0, st>>>(zn, wvt, nullptr, vb, 128, 128);
  k_gemm_bf16<1><<<dim3((unsigned)(R / 64), 2), 256, 0, st>>>(zn, wgatet, nullptr, gb, 128, 128);
  k_rowvec<<<dim3((unsigned)R), 128, 0, st>>>(zn, w_b, bias4, 128, 4);
  k_tri_flash<<<dim3(24, 4, 384), 32, 0, st>>>(qb, kb, vb, gb, bias4, mask, zn, Nn, trans);
  k_gemm_addf32<<<dim3((unsigned)(R / 64), 2), 256, 0, st>>>(zn, wot, b_o, Z, 128, 128);
}

static void transition_pass(float* X, const float* ln_g, const float* ln_b,
                            const bf16_t* wat, const bf16_t* wbt, const bf16_t* woutt,
                            bf16_t* xn, bf16_t* hid, size_t rows, int width,
                            int hidden, size_t chunk, hipStream_t st) {
  k_ln_rows<<<dim3((unsigned)rows), 256, 0, st>>>(X, ln_g, ln_b, xn, width);
  for (size_t r0 = 0; r0 < rows; r0 += chunk) {
    size_t cr = rows - r0 < chunk ? rows - r0 : chunk;
    k_dual_gate<2, 0><<<dim3((unsigned)(cr / 64), hidden / 64), 256, 0, st>>>(
        xn + r0 * width, wat, wbt, nullptr, hid, width, hidden, cr);
    k_gemm_addf32<<<dim3((unsigned)(cr / 64), width / 64), 256, 0, st>>>(
        hid, woutt, nullptr, X + r0 * width, hidden, width);
  }
}

extern "C" void kernel_launch(void* const* d_in, const int* in_sizes, int n_in,
                              void* d_out, int out_size, void* d_ws, size_t ws_size,
                              hipStream_t stream) {
  const int Nn = 384;
  const size_t R = (size_t)Nn * Nn;  // 147456
  auto f = [&](int i) { return (const float*)d_in[i]; };
  const float* s_in = f(0);
  const float* z_in = f(1);
  const float* mask = f(2);
  int p = 3;
  // params flattened in JAX pytree (sorted-key) order
  const float *apb_b_o = f(p + 0), *apb_b_qkv = f(p + 1), *apb_ln_s_b = f(p + 2),
              *apb_ln_s_g = f(p + 3), *apb_ln_z_b = f(p + 4), *apb_ln_z_g = f(p + 5),
              *apb_w_g = f(p + 6), *apb_w_o = f(p + 7), *apb_w_qkv = f(p + 8),
              *apb_w_zb = f(p + 9);
  p += 10;
  const float *pt_ln_b = f(p), *pt_ln_g = f(p + 1), *pt_w_a = f(p + 2),
              *pt_w_b = f(p + 3), *pt_w_out = f(p + 4);
  p += 5;
  const float *st_ln_b = f(p), *st_ln_g = f(p + 1), *st_w_a = f(p + 2),
              *st_w_b = f(p + 3), *st_w_out = f(p + 4);
  p += 5;
  const float *tae_b_o = f(p), *tae_ln_b = f(p + 1), *tae_ln_g = f(p + 2),
              *tae_w_b = f(p + 3), *tae_w_gate = f(p + 4), *tae_w_k = f(p + 5),
              *tae_w_o = f(p + 6), *tae_w_q = f(p + 7), *tae_w_v = f(p + 8);
  p += 9;
  const float *tas_b_o = f(p), *tas_ln_b = f(p + 1), *tas_ln_g = f(p + 2),
              *tas_w_b = f(p + 3), *tas_w_gate = f(p + 4), *tas_w_k = f(p + 5),
              *tas_w_o = f(p + 6), *tas_w_q = f(p + 7), *tas_w_v = f(p + 8);
  p += 9;
  const float *tmi_ln_in_b = f(p), *tmi_ln_in_g = f(p + 1), *tmi_ln_out_b = f(p + 2),
              *tmi_ln_out_g = f(p + 3), *tmi_w_ag = f(p + 4), *tmi_w_ap = f(p + 5),
              *tmi_w_bg = f(p + 6), *tmi_w_bp = f(p + 7), *tmi_w_g = f(p + 8),
              *tmi_w_z = f(p + 9);
  p += 10;
  const float *tmo_ln_in_b = f(p), *tmo_ln_in_g = f(p + 1), *tmo_ln_out_b = f(p + 2),
              *tmo_ln_out_g = f(p + 3), *tmo_w_ag = f(p + 4), *tmo_w_ap = f(p + 5),
              *tmo_w_bg = f(p + 6), *tmo_w_bp = f(p + 7), *tmo_w_g = f(p + 8),
              *tmo_w_z = f(p + 9);

  // workspace carving
  char* ws = (char*)d_ws;
  size_t off = 0;
  auto take = [&](size_t bytes) {
    size_t o = off;
    off += (bytes + 255) & ~(size_t)255;
    return o;
  };
  float* Z = (float*)(ws + take(R * 128 * 4));
  float* S = (float*)(ws + take(R * 4));
  bf16_t* B0 = (bf16_t*)(ws + take(R * 128 * 2));
  bf16_t* B1 = (bf16_t*)(ws + take(R * 128 * 2));
  bf16_t* B2 = (bf16_t*)(ws + take(R * 128 * 2));
  bf16_t* B3 = (bf16_t*)(ws + take(R * 128 * 2));
  bf16_t* B4 = (bf16_t*)(ws + take(R * 128 * 2));
  char*   B5 = ws + take((R / 4) * 512 * 2);

  // weight prep: f32 [K][N] -> bf16 [N][K]
  auto prep = [&](const float* W, int K, int Ncol) {
    bf16_t* wt = (bf16_t*)(ws + take((size_t)K * Ncol * 2));
    int tot = K * Ncol;
    k_wt<<<dim3((tot + 255) / 256), 256, 0, stream>>>(W, wt, K, Ncol);
    return wt;
  };
  const bf16_t *tmo_wag_t = prep(tmo_w_ag, 128, 128), *tmo_wap_t = prep(tmo_w_ap, 128, 128),
               *tmo_wbg_t = prep(tmo_w_bg, 128, 128), *tmo_wbp_t = prep(tmo_w_bp, 128, 128),
               *tmo_wz_t = prep(tmo_w_z, 128, 128), *tmo_wg_t = prep(tmo_w_g, 128, 128);
  const bf16_t *tmi_wag_t = prep(tmi_w_ag, 128, 128), *tmi_wap_t = prep(tmi_w_ap, 128, 128),
               *tmi_wbg_t = prep(tmi_w_bg, 128, 128), *tmi_wbp_t = prep(tmi_w_bp, 128, 128),
               *tmi_wz_t = prep(tmi_w_z, 128, 128), *tmi_wg_t = prep(tmi_w_g, 128, 128);
  const bf16_t *tas_wq_t = prep(tas_w_q, 128, 128), *tas_wk_t = prep(tas_w_k, 128, 128),
               *tas_wv_t = prep(tas_w_v, 128, 128), *tas_wgate_t = prep(tas_w_gate, 128, 128),
               *tas_wo_t = prep(tas_w_o, 128, 128);
  const bf16_t *tae_wq_t = prep(tae_w_q, 128, 128), *tae_wk_t = prep(tae_w_k, 128, 128),
               *tae_wv_t = prep(tae_w_v, 128, 128), *tae_wgate_t = prep(tae_w_gate, 128, 128),
               *tae_wo_t = prep(tae_w_o, 128, 128);
  const bf16_t *pt_wa_t = prep(pt_w_a, 128, 512), *pt_wb_t = prep(pt_w_b, 128, 512),
               *pt_wout_t = prep(pt_w_out, 512, 128);
  const bf16_t *apb_wqkv_t = prep(apb_w_qkv, 384, 1152), *apb_wg_t = prep(apb_w_g, 384, 384),
               *apb_wo_t = prep(apb_w_o, 384, 384);
  const bf16_t *st_wa_t = prep(st_w_a, 384, 1536), *st_wb_t = prep(st_w_b, 384, 1536),
               *st_wout_t = prep(st_w_out, 1536, 384);

  hipMemcpyAsync(Z, z_in, R * 128 * 4, hipMemcpyDeviceToDevice, stream);
  hipMemcpyAsync(S, s_in, R * 4, hipMemcpyDeviceToDevice, stream);

  // 1) triangle multiplication, outgoing then incoming
  tri_mul_pass(Z, mask, tmo_ln_in_g, tmo_ln_in_b, tmo_wag_t, tmo_wap_t, tmo_wbg_t,
               tmo_wbp_t, tmo_ln_out_g, tmo_ln_out_b, tmo_wz_t, tmo_wg_t,
               B0, B1, B2, B3, B4, 1, stream);
  tri_mul_pass(Z, mask, tmi_ln_in_g, tmi_ln_in_b, tmi_wag_t, tmi_wap_t, tmi_wbg_t,
               tmi_wbp_t, tmi_ln_out_g, tmi_ln_out_b, tmi_wz_t, tmi_wg_t,
               B0, B1, B2, B3, B4, 0, stream);

  // 2) triangle attention, starting (trans=0) then ending (trans=1)
  tri_att_pass(Z, mask, tas_ln_g, tas_ln_b, tas_wq_t, tas_wk_t, tas_wv_t, tas_w_b,
               tas_wgate_t, tas_wo_t, tas_b_o, B0, B1, B2, B3, B4, (float*)B5, 0, stream);
  tri_att_pass(Z, mask, tae_ln_g, tae_ln_b, tae_wq_t, tae_wk_t, tae_wv_t, tae_w_b,
               tae_wgate_t, tae_wo_t, tae_b_o, B0, B1, B2, B3, B4, (float*)B5, 1, stream);

  // 3) pair transition (chunked hidden to bound workspace)
  transition_pass(Z, pt_ln_g, pt_ln_b, pt_wa_t, pt_wb_t, pt_wout_t,
                  B0, (bf16_t*)B5, R, 128, 512, R / 4, stream);

  // 4) pair-bias attention on s
  {
    bf16_t* SN = B1;
    bf16_t* QKV = B2;
    float* BIAS16 = (float*)B5;
    bf16_t* SGATE = B4;
    bf16_t* SOUT = B0;
    k_ln_rows<<<dim3(Nn), 256, 0, stream>>>(S, apb_ln_s_g, apb_ln_s_b, SN, 384);
    k_gemm_bf16<0><<<dim3(6, 18), 256, 0, stream>>>(SN, apb_wqkv_t, apb_b_qkv, QKV, 384, 1152);
    k_ln_rows<<<dim3((unsigned)R), 256, 0, stream>>>(Z, apb_ln_z_g, apb_ln_z_b, B0, 128);
    k_rowvec<<<dim3((unsigned)R), 128, 0, stream>>>(B0, apb_w_zb, BIAS16, 128, 16);
    k_gemm_bf16<1><<<dim3(6, 6), 256, 0, stream>>>(SN, apb_wg_t, nullptr, SGATE, 384, 384);
    k_s_flash<<<dim3(24, 16), 32, 0, stream>>>(QKV, BIAS16, mask, SGATE, SOUT, Nn);
    k_gemm_addf32<<<dim3(6, 6), 256, 0, stream>>>(SOUT, apb_wo_t, apb_b_o, S, 384, 384);
  }

  // 5) single transition
  transition_pass(S, st_ln_g, st_ln_b, st_wa_t, st_wb_t, st_wout_t,
                  B1, (bf16_t*)B5, 384, 384, 1536, 384, stream);

  // outputs: (s, z) concatenated, f32
  float* outf = (float*)d_out;
  hipMemcpyAsync(outf, S, R * 4, hipMemcpyDeviceToDevice, stream);
  hipMemcpyAsync(outf + R, Z, R * 128 * 4, hipMemcpyDeviceToDevice, stream);
}
